// KlearAttention_43963285242587
// MI455X (gfx1250) — matrix operand order repaired and checked by reading, not compiled, not run on hardware
//
#include <hip/hip_runtime.h>

// ---------------------------------------------------------------- constants
#define HH   32          // query heads
#define KHH  8           // kv heads
#define HD_  128         // head dim
#define BB   2
#define LL   2048
#define DD   4096
#define MM   (BB*LL)     // 4096 rows
#define QB   64          // queries per attention workgroup
#define KB   32          // keys / K-slab per iteration

typedef __attribute__((ext_vector_type(16))) __bf16 v16bf;
typedef __attribute__((ext_vector_type(8)))  float  v8f;

union Frag { uint4 q[2]; v16bf v; };

#ifndef __has_builtin
#define __has_builtin(x) 0
#endif

#if __has_builtin(__builtin_amdgcn_tensor_load_to_lds) && \
    __has_builtin(__builtin_amdgcn_s_wait_tensorcnt)
#define USE_TDM 1
#else
#define USE_TDM 0
#endif

#if __has_builtin(__builtin_amdgcn_global_load_async_to_lds_b128) && \
    __has_builtin(__builtin_amdgcn_s_wait_asynccnt)
#define USE_ASYNC 1
#else
#define USE_ASYNC 0
#endif

#if USE_ASYNC
typedef int v4i_vs __attribute__((__vector_size__(16)));
typedef __attribute__((address_space(1))) v4i_vs* g_v4i_ptr;
typedef __attribute__((address_space(3))) v4i_vs* l_v4i_ptr;
#endif

// 16-byte global -> LDS copy (async on CDNA5 if available)
__device__ __forceinline__ void copy16(const unsigned short* g, unsigned short* l) {
#if USE_ASYNC
    __builtin_amdgcn_global_load_async_to_lds_b128(
        (g_v4i_ptr)(unsigned short*)g, (l_v4i_ptr)l, 0, 0);
#else
    *reinterpret_cast<uint4*>(l) = *reinterpret_cast<const uint4*>(g);
#endif
}

#if USE_TDM
typedef unsigned int u32x4_vs __attribute__((__vector_size__(16)));
typedef int          i32x8_vs __attribute__((__vector_size__(32)));
typedef int          i32x4_vs __attribute__((__vector_size__(16)));

// Tensor Data Mover: 2D tile of bf16 (tile_w x tile_h elements, row stride
// in elements) from global -> LDS, contiguous in LDS. One instruction.
__device__ __forceinline__ void tdm_load_2d(const unsigned short* g,
                                            unsigned short* l,
                                            unsigned int tile_w,
                                            unsigned int tile_h,
                                            unsigned int row_stride) {
    unsigned long long ga = (unsigned long long)g;
    unsigned int loff = (unsigned int)(unsigned long long)
        (__attribute__((address_space(3))) unsigned short*)l;
    // D# group0: count=1 | lds_addr | global_addr[56:0] | type=2
    u32x4_vs g0 = { 1u, loff, (unsigned int)ga,
                    ((unsigned int)(ga >> 32) & 0x01FFFFFFu) | 0x80000000u };
    // D# group1: data_size=1(2B); tensor_dim0=tile_w; tensor_dim1=tile_h;
    //            tile_dim0=tile_w; tile_dim1=tile_h; dim0_stride=row_stride
    i32x8_vs g1 = { (int)0x00010000u,
                    (int)(tile_w << 16),
                    (int)((tile_w >> 16) | (tile_h << 16)),
                    (int)(tile_w << 16),
                    (int)tile_h,
                    (int)row_stride,
                    0, 0 };
    i32x4_vs z4 = { 0, 0, 0, 0 };
    i32x8_vs z8 = { 0, 0, 0, 0, 0, 0, 0, 0 };
    __builtin_amdgcn_tensor_load_to_lds(g0, g1, z4, z4, z8, 0);
}
#endif

// wait for staged tile data (mode-dependent); call before __syncthreads()
__device__ __forceinline__ void stage_wait(int wave) {
#if USE_TDM
    if (wave == 0) __builtin_amdgcn_s_wait_tensorcnt(0);
#elif USE_ASYNC
    __builtin_amdgcn_s_wait_asynccnt(0);
#endif
    (void)wave;
}

__device__ __forceinline__ unsigned short f2bf(float f) {
    unsigned int u = __float_as_uint(f);
    u += 0x7fffu + ((u >> 16) & 1u);
    return (unsigned short)(u >> 16);
}
__device__ __forceinline__ float bf2f(unsigned short h) {
    return __uint_as_float(((unsigned int)h) << 16);
}
__device__ __forceinline__ v8f vzero() {
    v8f z;
#pragma unroll
    for (int i = 0; i < 8; ++i) z[i] = 0.0f;
    return z;
}
__device__ __forceinline__ v8f wmma_bf16(v16bf a, v16bf b, v8f c) {
    return __builtin_amdgcn_wmma_f32_16x16x32_bf16(false, a, false, b, (short)0, c, false, false);
}

// ---------------------------------------------------------------- fp32 -> bf16 (row-major)
__global__ __launch_bounds__(256) void cvt_bf16(const float* __restrict__ in,
                                                unsigned short* __restrict__ out,
                                                long n) {
    long i = (long)blockIdx.x * 256 + threadIdx.x;
    long stride = (long)gridDim.x * 256;
    for (; i < n; i += stride) out[i] = f2bf(in[i]);
}

// ---------------------------------------------------------------- fp32 KxN -> bf16 NxK (transpose)
__global__ __launch_bounds__(256) void cvt_transpose(const float* __restrict__ in,
                                                     unsigned short* __restrict__ out,
                                                     int K, int N) {
    __shared__ unsigned short t[32][33];
    int k0 = blockIdx.y * 32, n0 = blockIdx.x * 32;
    int lx = threadIdx.x, ly = threadIdx.y;           // 32 x 8
#pragma unroll
    for (int j = 0; j < 4; ++j) {
        int k = k0 + ly + j * 8;
        t[ly + j * 8][lx] = f2bf(in[(size_t)k * N + n0 + lx]);
    }
    __syncthreads();
#pragma unroll
    for (int j = 0; j < 4; ++j) {
        int n = n0 + ly + j * 8;
        out[(size_t)n * K + k0 + lx] = t[lx][ly + j * 8];
    }
}

// ---------------------------------------------------------------- bf16 V transpose
// in: (B, L, KH, HD)   out: (B, KH, HD, L)
__global__ __launch_bounds__(256) void transpose_v(const unsigned short* __restrict__ in,
                                                   unsigned short* __restrict__ out) {
    __shared__ unsigned short t[32][33];
    int l0 = blockIdx.x * 32, d0 = blockIdx.y * 32;
    int b = blockIdx.z / KHH, kvh = blockIdx.z % KHH;
    int lx = threadIdx.x, ly = threadIdx.y;
#pragma unroll
    for (int j = 0; j < 4; ++j) {
        int l = l0 + ly + j * 8;
        t[ly + j * 8][lx] = in[(((size_t)b * LL + l) * KHH + kvh) * HD_ + d0 + lx];
    }
    __syncthreads();
#pragma unroll
    for (int j = 0; j < 4; ++j) {
        int d = d0 + ly + j * 8;
        out[(((size_t)b * KHH + kvh) * HD_ + d) * LL + l0 + lx] = t[lx][ly + j * 8];
    }
}

// ---------------------------------------------------------------- WMMA GEMM (A * Bt^T)
// C[M,N] = A[M,K] (bf16 row-major) * Bt[N,K]^T (bf16, pre-transposed weights)
// 128x128 block tile, 256 thr = 8 waves, wave tile 32x64; double-buffered LDS,
// TDM (or async) global->LDS staging overlapped with WMMA.
template <bool OUT_F32>
__global__ __launch_bounds__(256) void gemm_nt(const unsigned short* __restrict__ A,
                                               const unsigned short* __restrict__ Bt,
                                               void* __restrict__ Cv,
                                               int M, int N, int K) {
    __shared__ alignas(16) unsigned short As[2][128 * 32];   // [row][k]
    __shared__ alignas(16) unsigned short Bs[2][128 * 32];   // [col][k]

    const int tid   = threadIdx.x;
    const int wave  = tid >> 5;
    const int lane  = tid & 31;
    const int lr    = lane & 15;
    const int hs    = lane >> 4;
    const int waveM = wave & 3;
    const int waveN = wave >> 2;
    const int rowBase = blockIdx.y * 128;
    const int colBase = blockIdx.x * 128;

    const int r    = tid >> 1;              // staging row/col 0..127
    const int kOff = (tid & 1) * 16;        // staging half-row offset

    v8f acc[2][4];
#pragma unroll
    for (int m = 0; m < 2; ++m)
#pragma unroll
        for (int n = 0; n < 4; ++n) acc[m][n] = vzero();

    auto stage = [&](int kt, int buf) {
#if USE_TDM
        if (wave == 0) {
            tdm_load_2d(A  + (size_t)rowBase * K + (size_t)kt * KB, &As[buf][0], KB, 128, K);
            tdm_load_2d(Bt + (size_t)colBase * K + (size_t)kt * KB, &Bs[buf][0], KB, 128, K);
        }
#else
        const unsigned short* ga = A  + (size_t)(rowBase + r) * K + (size_t)kt * KB + kOff;
        const unsigned short* gb = Bt + (size_t)(colBase + r) * K + (size_t)kt * KB + kOff;
        copy16(ga,     &As[buf][r * 32 + kOff]);
        copy16(ga + 8, &As[buf][r * 32 + kOff + 8]);
        copy16(gb,     &Bs[buf][r * 32 + kOff]);
        copy16(gb + 8, &Bs[buf][r * 32 + kOff + 8]);
#if !USE_ASYNC
        __builtin_prefetch(ga + KB, 0, 3);
        __builtin_prefetch(gb + KB, 0, 3);
#endif
#endif
    };

    const int nk = K / KB;
    stage(0, 0);
    for (int kt = 0; kt < nk; ++kt) {
        const int cur = kt & 1;
        stage_wait(wave);
        __syncthreads();
        if (kt + 1 < nk) stage(kt + 1, cur ^ 1);

        Frag af[2], bfg[4];
#pragma unroll
        for (int m = 0; m < 2; ++m) {
            int row = waveM * 32 + m * 16 + lr;
            af[m].q[0] = *reinterpret_cast<const uint4*>(&As[cur][row * 32 + hs * 16]);
            af[m].q[1] = *reinterpret_cast<const uint4*>(&As[cur][row * 32 + hs * 16 + 8]);
        }
#pragma unroll
        for (int n = 0; n < 4; ++n) {
            int col = waveN * 64 + n * 16 + lr;
            const uint4* p = reinterpret_cast<const uint4*>(&Bs[cur][col * 32 + hs * 16]);
            bfg[n].q[0] = p[0];
            bfg[n].q[1] = p[1];
        }
#pragma unroll
        for (int m = 0; m < 2; ++m)
#pragma unroll
            for (int n = 0; n < 4; ++n)
                acc[m][n] = wmma_bf16(af[m].v, bfg[n].v, acc[m][n]);
    }

    // epilogue: C layout lane -> row = r + 8*hs, col = lr
#pragma unroll
    for (int m = 0; m < 2; ++m)
#pragma unroll
        for (int n = 0; n < 4; ++n)
#pragma unroll
            for (int rr = 0; rr < 8; ++rr) {
                int row = rowBase + waveM * 32 + m * 16 + rr + 8 * hs;
                int col = colBase + waveN * 64 + n * 16 + lr;
                float val = acc[m][n][rr];
                if (OUT_F32)
                    reinterpret_cast<float*>(Cv)[(size_t)row * N + col] = val;
                else
                    reinterpret_cast<unsigned short*>(Cv)[(size_t)row * N + col] = f2bf(val);
            }
}

// ---------------------------------------------------------------- RMSNorm + RoPE
// in : (B, L, NH, HD) bf16    out: (B, NH, L, HD) bf16
__global__ __launch_bounds__(128) void normrope(const unsigned short* __restrict__ in,
                                                unsigned short* __restrict__ out,
                                                const float* __restrict__ w,
                                                int NH) {
    __shared__ float red[4];
    __shared__ float nv[HD_];

    int bid = blockIdx.x;
    int h = bid % NH;
    int l = (bid / NH) % LL;
    int b = bid / (NH * LL);
    int d = threadIdx.x;
    int wv = d >> 5;

    size_t ibase = (((size_t)b * LL + l) * NH + h) * HD_;
    size_t obase = (((size_t)b * NH + h) * LL + l) * HD_;

    float v = bf2f(in[ibase + d]);
    float s = v * v;
#pragma unroll
    for (int off = 16; off > 0; off >>= 1) s += __shfl_xor(s, off, 32);
    if ((d & 31) == 0) red[wv] = s;
    __syncthreads();
    float mean = (red[0] + red[1] + red[2] + red[3]) * (1.0f / HD_);
    float rs = rsqrtf(mean + 1e-5f);
    nv[d] = v * rs * w[d];
    __syncthreads();

    int j = d & 63;
    float inv_freq = __expf(-(float)j * (13.815510557964274f / 64.0f)); // 1e6^(-j/64)
    float ang = (float)l * inv_freq;
    float c, sn;
    __sincosf(ang, &sn, &c);
    float x1 = nv[j], x2 = nv[j + 64];
    float o = (d < 64) ? (x1 * c - x2 * sn) : (x2 * c + x1 * sn);
    out[obase + d] = f2bf(o);
}

// ---------------------------------------------------------------- flash attention
// qh: (B,H,L,HD)  kh: (B,KH,L,HD)  vT: (B,KH,HD,L)  out: (B,L,H*HD)  all bf16
__global__ __launch_bounds__(128) void attn_fwd(const unsigned short* __restrict__ qh,
                                                const unsigned short* __restrict__ kh,
                                                const unsigned short* __restrict__ vT,
                                                unsigned short* __restrict__ aout) {
    __shared__ alignas(16) unsigned short ks[2][KB * HD_];   // [key][d]
    __shared__ alignas(16) unsigned short vt[2][HD_ * KB];   // [d][key]
    __shared__ alignas(16) unsigned short ps[4][16 * 32];    // per-wave P tile

    const int qblk = blockIdx.x;
    const int h    = blockIdx.y;
    const int b    = blockIdx.z;
    const int kvh  = h >> 2;                 // H/KH = 4
    const int tid  = threadIdx.x;
    const int wave = tid >> 5;
    const int lane = tid & 31;
    const int lr   = lane & 15;
    const int hs   = lane >> 4;

    const int qrow0 = qblk * QB + wave * 16;
    const size_t qbase = (((size_t)b * HH + h) * LL + qrow0) * HD_;
    const size_t kbase = ((size_t)b * KHH + kvh) * LL * HD_;
    const size_t vbase = ((size_t)b * KHH + kvh) * (size_t)HD_ * LL;

    // preload Q fragments (16 rows x 128 dims = 4 k-steps)
    Frag qf[4];
#pragma unroll
    for (int kk = 0; kk < 4; ++kk) {
        const unsigned short* qp = qh + qbase + (size_t)lr * HD_ + kk * 32 + hs * 16;
        qf[kk].q[0] = *reinterpret_cast<const uint4*>(qp);
        qf[kk].q[1] = *reinterpret_cast<const uint4*>(qp + 8);
    }

    v8f O[8];
#pragma unroll
    for (int t = 0; t < 8; ++t) O[t] = vzero();
    float mrow[8], lrow[8];
#pragma unroll
    for (int r = 0; r < 8; ++r) { mrow[r] = -1e30f; lrow[r] = 0.0f; }

    auto stage = [&](int kb, int buf) {
        const int key0 = kb * KB;
#if USE_TDM
        if (wave == 0) {
            tdm_load_2d(kh + kbase + (size_t)key0 * HD_, &ks[buf][0], HD_, KB, HD_);
            tdm_load_2d(vT + vbase + key0,               &vt[buf][0], KB, HD_, LL);
        }
#else
        // K block: 512 chunks of 8 halves, coalesced
#pragma unroll
        for (int it = 0; it < 4; ++it) {
            int c = tid + it * 128;
            int key = c >> 4;
            int dO  = (c & 15) * 8;
            copy16(kh + kbase + (size_t)(key0 + key) * HD_ + dO,
                   &ks[buf][key * HD_ + dO]);
        }
        // V block from (HD, L) layout: contiguous along keys
#pragma unroll
        for (int it = 0; it < 4; ++it) {
            int c = tid + it * 128;
            int d   = c >> 2;
            int kO  = (c & 3) * 8;
            copy16(vT + vbase + (size_t)d * LL + key0 + kO,
                   &vt[buf][d * KB + kO]);
        }
#endif
    };

    const float scale = 0.08838834764831845f;   // 128^-0.5
    const int nkb = (qblk * QB + QB) / KB;      // causal

    stage(0, 0);
    for (int kb = 0; kb < nkb; ++kb) {
        const int cur = kb & 1;
        stage_wait(wave);
        __syncthreads();
        if (kb + 1 < nkb) stage(kb + 1, cur ^ 1);

        const int key0 = kb * KB;
        // ---- scores S = Q @ K^T
        v8f s0 = vzero(), s1 = vzero();
#pragma unroll
        for (int kk = 0; kk < 4; ++kk) {
            Frag bk0, bk1;
            const uint4* p0 = reinterpret_cast<const uint4*>(&ks[cur][lr * HD_ + kk * 32 + hs * 16]);
            bk0.q[0] = p0[0]; bk0.q[1] = p0[1];
            const uint4* p1 = reinterpret_cast<const uint4*>(&ks[cur][(16 + lr) * HD_ + kk * 32 + hs * 16]);
            bk1.q[0] = p1[0]; bk1.q[1] = p1[1];
            s0 = wmma_bf16(qf[kk].v, bk0.v, s0);
            s1 = wmma_bf16(qf[kk].v, bk1.v, s1);
        }

        // ---- online softmax (row = r + 8*hs, N = lr)
#pragma unroll
        for (int r = 0; r < 8; ++r) {
            int qi  = qrow0 + r + 8 * hs;
            float a0 = s0[r] * scale + (((key0 + lr) <= qi) ? 0.0f : -1e9f);
            float a1 = s1[r] * scale + (((key0 + 16 + lr) <= qi) ? 0.0f : -1e9f);
            float mx = fmaxf(a0, a1);
#pragma unroll
            for (int off = 8; off > 0; off >>= 1) mx = fmaxf(mx, __shfl_xor(mx, off, 32));
            float nm   = fmaxf(mrow[r], mx);
            float corr = __expf(mrow[r] - nm);
            mrow[r] = nm;
            float p0v = __expf(a0 - nm);
            float p1v = __expf(a1 - nm);
            float rsum = p0v + p1v;
#pragma unroll
            for (int off = 8; off > 0; off >>= 1) rsum += __shfl_xor(rsum, off, 32);
            lrow[r] = lrow[r] * corr + rsum;
#pragma unroll
            for (int t = 0; t < 8; ++t) O[t][r] *= corr;
            ps[wave][(r + 8 * hs) * 32 + lr]      = f2bf(p0v);
            ps[wave][(r + 8 * hs) * 32 + 16 + lr] = f2bf(p1v);
        }

        // ---- reload P as A-fragment
        Frag pf;
        const unsigned short* pp = &ps[wave][lr * 32 + hs * 16];
        pf.q[0] = *reinterpret_cast<const uint4*>(pp);
        pf.q[1] = *reinterpret_cast<const uint4*>(pp + 8);

        // ---- O += P @ V
#pragma unroll
        for (int t = 0; t < 8; ++t) {
            Frag vf;
            const uint4* vp = reinterpret_cast<const uint4*>(&vt[cur][(t * 16 + lr) * KB + hs * 16]);
            vf.q[0] = vp[0]; vf.q[1] = vp[1];
            O[t] = wmma_bf16(pf.v, vf.v, O[t]);
        }
    }

    // ---- normalize + store bf16 (B, L, H*HD)
#pragma unroll
    for (int t = 0; t < 8; ++t)
#pragma unroll
        for (int r = 0; r < 8; ++r) {
            int qi  = qrow0 + r + 8 * hs;
            int col = h * HD_ + t * 16 + lr;
            float o = O[t][r] / lrow[r];
            aout[((size_t)b * LL + qi) * (HH * HD_) + col] = f2bf(o);
        }
}

// ---------------------------------------------------------------- launcher
extern "C" void kernel_launch(void* const* d_in, const int* in_sizes, int n_in,
                              void* d_out, int out_size, void* d_ws, size_t ws_size,
                              hipStream_t stream) {
    const float* x   = (const float*)d_in[0];
    const float* wq  = (const float*)d_in[1];
    const float* wk  = (const float*)d_in[2];
    const float* wv  = (const float*)d_in[3];
    const float* wo  = (const float*)d_in[4];
    const float* qnw = (const float*)d_in[5];
    const float* knw = (const float*)d_in[6];
    // d_in[7] (mask) unused: causal mask applied analytically.

    char* ws = (char*)d_ws;
    size_t off = 0;
    auto alloc = [&](size_t bytes) { size_t r = off; off += (bytes + 255) & ~(size_t)255; return r; };

    unsigned short* xb  = (unsigned short*)(ws + alloc((size_t)MM * DD * 2));
    unsigned short* wqT = (unsigned short*)(ws + alloc((size_t)DD * 4096 * 2));   // (N=4096, K=4096)
    unsigned short* wkT = (unsigned short*)(ws + alloc((size_t)DD * 1024 * 2));   // (1024, 4096)
    unsigned short* wvT = (unsigned short*)(ws + alloc((size_t)DD * 1024 * 2));
    unsigned short* woT = (unsigned short*)(ws + alloc((size_t)DD * 4096 * 2));   // (4096, 4096)
    unsigned short* qb  = (unsigned short*)(ws + alloc((size_t)MM * 4096 * 2));
    unsigned short* kbX = (unsigned short*)(ws + alloc((size_t)MM * 1024 * 2));
    unsigned short* vbX = (unsigned short*)(ws + alloc((size_t)MM * 1024 * 2));
    unsigned short* qhb = (unsigned short*)(ws + alloc((size_t)MM * 4096 * 2));
    unsigned short* khb = (unsigned short*)(ws + alloc((size_t)MM * 1024 * 2));
    unsigned short* vtg = (unsigned short*)(ws + alloc((size_t)MM * 1024 * 2));
    unsigned short* atb = (unsigned short*)(ws + alloc((size_t)MM * 4096 * 2));

    // convert activations (row-major), transpose-convert weights (NxK)
    {
        long n = (long)MM * DD;
        cvt_bf16<<<(int)((n + 255) / 256), 256, 0, stream>>>(x, xb, n);
    }
    cvt_transpose<<<dim3(4096 / 32, DD / 32), dim3(32, 8), 0, stream>>>(wq, wqT, DD, 4096);
    cvt_transpose<<<dim3(1024 / 32, DD / 32), dim3(32, 8), 0, stream>>>(wk, wkT, DD, 1024);
    cvt_transpose<<<dim3(1024 / 32, DD / 32), dim3(32, 8), 0, stream>>>(wv, wvT, DD, 1024);
    cvt_transpose<<<dim3(4096 / 32, DD / 32), dim3(32, 8), 0, stream>>>(wo, woT, DD, 4096);

    // QKV projections (bf16 out)
    gemm_nt<false><<<dim3(4096 / 128, MM / 128), 256, 0, stream>>>(xb, wqT, qb, MM, 4096, DD);
    gemm_nt<false><<<dim3(1024 / 128, MM / 128), 256, 0, stream>>>(xb, wkT, kbX, MM, 1024, DD);
    gemm_nt<false><<<dim3(1024 / 128, MM / 128), 256, 0, stream>>>(xb, wvT, vbX, MM, 1024, DD);

    // RMSNorm + RoPE to head-major; V to (B,KH,HD,L)
    normrope<<<BB * LL * HH,  128, 0, stream>>>(qb,  qhb, qnw, HH);
    normrope<<<BB * LL * KHH, 128, 0, stream>>>(kbX, khb, knw, KHH);
    transpose_v<<<dim3(LL / 32, HD_ / 32, BB * KHH), dim3(32, 8), 0, stream>>>(vbX, vtg);

    // flash attention
    attn_fwd<<<dim3(LL / QB, HH, BB), 128, 0, stream>>>(qhb, khb, vtg, atb);

    // output projection (fp32 out)
    gemm_nt<true><<<dim3(4096 / 128, MM / 128), 256, 0, stream>>>(atb, woT, (float*)d_out, MM, 4096, DD);

    (void)in_sizes; (void)n_in; (void)out_size; (void)ws_size;
}
